// OverlappingMaskEstimationModule_55825984913602
// MI455X (gfx1250) — compile-verified, matrix operand order checked by reading, not run on hardware
//
#include <hip/hip_runtime.h>
#include <hip/hip_bf16.h>
#include <math.h>

typedef __attribute__((ext_vector_type(16))) __bf16 v16bf;
typedef __attribute__((ext_vector_type(8)))  float  v8f;

#define N_BANDS 15
#define BW      129
#define HOP     64
#define N_FREQ  1025
#define EMB     128
#define MLP     512
#define OUT_    516           // BW*C*REIM
#define TWOOUT  1032
#define TTILE   128

// LDS layout
#define AU_STRIDE_U32 68      // padded row stride for qn pairs (16B-aligned, bank-spread)
#define H_STRIDE_H    520     // padded row stride for H (bf16); 260 u32 pairs
#define BS_STRIDE_U32 260     // per-column pair stride for transposed B tiles
#define OFF_AU   0
#define OFF_H    (128*AU_STRIDE_U32*4)                   // 34816
#define OFF_BSA  (OFF_H + 128*H_STRIDE_H*2)              // 34816+133120 = 167936
#define OFF_BSG  (OFF_BSA + 16*BS_STRIDE_U32*4)          // +16640 = 184576
#define SMEM_BYTES (OFF_BSG + 16*BS_STRIDE_U32*4)        // 201216

__device__ __forceinline__ unsigned f2bf1(float x) {
    unsigned u = __builtin_bit_cast(unsigned, x);
    u += 0x7FFFu + ((u >> 16) & 1u);          // round-to-nearest-even
    return u >> 16;
}
__device__ __forceinline__ unsigned packbf(float a, float b) {
    return f2bf1(a) | (f2bf1(b) << 16);
}
// branchless tanh: v_exp_f32 + v_rcp_f32, no divergence, no IEEE div sequence
__device__ __forceinline__ float fast_tanh(float x) {
    float ax = __builtin_fabsf(x);
    float e  = __expf(-2.0f * ax);
    float r  = (1.0f - e) * __builtin_amdgcn_rcpf(1.0f + e);
    return __builtin_copysignf(r, x);
}
__device__ __forceinline__ float fast_sigmoid(float g) {
    return __builtin_amdgcn_rcpf(1.0f + __expf(-g));
}

union Frag { unsigned u[8]; uint4 q[2]; v16bf v; };

// lane's fragment = pairs [base .. base+3] and [base+8 .. base+11] (contiguous 16B chunks)
__device__ __forceinline__ void load_frag(Frag& f, const unsigned* basePtr) {
    const uint4* p = reinterpret_cast<const uint4*>(basePtr);
    f.q[0] = p[0];
    f.q[1] = p[2];
}

__global__ void zero_f32(float* __restrict__ p, long long n) {
    long long i = (long long)blockIdx.x * blockDim.x + threadIdx.x;
    long long stride = (long long)gridDim.x * blockDim.x;
    for (; i < n; i += stride) p[i] = 0.0f;
}

__global__ __launch_bounds__(256, 1)
void band_mlp_mask_kernel(const float* __restrict__ q,
                          const float* __restrict__ ln_gamma,
                          const float* __restrict__ ln_beta,
                          const float* __restrict__ W1,
                          const float* __restrict__ b1,
                          const float* __restrict__ W2,
                          const float* __restrict__ b2,
                          const float* __restrict__ fw,
                          float* __restrict__ outp)
{
    extern __shared__ __align__(16) char smem[];
    unsigned*       Au  = (unsigned*)(smem + OFF_AU);
    unsigned short* Hs  = (unsigned short*)(smem + OFF_H);
    unsigned*       Hu  = (unsigned*)(smem + OFF_H);
    unsigned*       BsA = (unsigned*)(smem + OFF_BSA);   // transposed: [N][pair]
    unsigned*       BsG = (unsigned*)(smem + OFF_BSG);

    const int tid  = threadIdx.x;
    const int w    = tid >> 5;       // wave id 0..7 (wave32)
    const int lid  = tid & 31;
    const int half = lid >> 4;
    const int l16  = lid & 15;
    const int m0   = w * 16;         // wave-owned M block

    int bx = blockIdx.x;
    const int n  = bx % N_BANDS; bx /= N_BANDS;
    const int b  = bx & 1;
    const int tt = bx >> 1;          // 0..15
    const int t0 = tt * TTILE;

    const float* qbase = q + (((long long)b * N_BANDS + n) * 2048 + t0) * EMB;
    const float* gmp = ln_gamma + n * EMB;
    const float* btp = ln_beta  + n * EMB;
    const float* W1p = W1 + (long long)n * EMB * MLP;
    const float* b1p = b1 + n * MLP;
    const float* W2p = W2 + (long long)n * MLP * TWOOUT;
    const float* b2p = b2 + n * TWOOUT;
    const float* fwp = fw + n * BW;

    // ---------- Phase 1: LayerNorm -> bf16 qn tile in LDS ----------
    float4 gv = reinterpret_cast<const float4*>(gmp)[lid];
    float4 bv = reinterpret_cast<const float4*>(btp)[lid];
    for (int i = 0; i < 16; ++i) {
        int row = m0 + i;
        float4 x = reinterpret_cast<const float4*>(qbase + (long long)row * EMB)[lid];
        float s  = x.x + x.y + x.z + x.w;
        float s2 = x.x * x.x + x.y * x.y + x.z * x.z + x.w * x.w;
        #pragma unroll
        for (int d = 16; d >= 1; d >>= 1) {
            s  += __shfl_xor(s,  d, 32);
            s2 += __shfl_xor(s2, d, 32);
        }
        float mean = s * (1.0f / 128.0f);
        float var  = s2 * (1.0f / 128.0f) - mean * mean;
        float rstd = rsqrtf(var + 1e-5f);
        float y0 = (x.x - mean) * rstd * gv.x + bv.x;
        float y1 = (x.y - mean) * rstd * gv.y + bv.y;
        float y2 = (x.z - mean) * rstd * gv.z + bv.z;
        float y3 = (x.w - mean) * rstd * gv.w + bv.w;
        Au[row * AU_STRIDE_U32 + lid * 2 + 0] = packbf(y0, y1);
        Au[row * AU_STRIDE_U32 + lid * 2 + 1] = packbf(y2, y3);
    }

    // ---------- Phase 2: GEMM1 (qn @ W1) + tanh -> H bf16 in LDS ----------
    for (int f0 = 0; f0 < MLP; f0 += 16) {
        __syncthreads();
        // stage 128x16 B tile of W1, transposed fragment-major: BsA[N][kpair]
        for (int i = tid; i < 1024; i += 256) {
            int kp = i >> 4, N = i & 15;
            int k = kp * 2;
            BsA[N * BS_STRIDE_U32 + kp] =
                packbf(W1p[(long long)k * MLP + f0 + N],
                       W1p[(long long)(k + 1) * MLP + f0 + N]);
        }
        __syncthreads();
        v8f acc = {};
        for (int k0 = 0; k0 < EMB; k0 += 32) {
            int pb = (k0 >> 1) + half * 4;
            Frag aF, bF;
            load_frag(aF, &Au[(m0 + l16) * AU_STRIDE_U32 + pb]);
            load_frag(bF, &BsA[l16 * BS_STRIDE_U32 + pb]);
            acc = __builtin_amdgcn_wmma_f32_16x16x32_bf16(
                false, aF.v, false, bF.v, (short)0, acc, false, false);
        }
        float bias = b1p[f0 + l16];
        #pragma unroll
        for (int r = 0; r < 8; ++r) {
            int m = m0 + r + half * 8;
            float h = fast_tanh(acc[r] + bias);
            Hs[m * H_STRIDE_H + f0 + l16] = (unsigned short)f2bf1(h);
        }
    }

    __syncthreads();

    // ---------- Phase 3: GEMM2 paired (a | g) column tiles + GLU + scatter ----------
    for (int jt = 0; jt < 33; ++jt) {       // 33*16 >= 516
        int j0 = jt * 16;
        __syncthreads();
        // stage two 512x16 B tiles (transposed): a-cols [j0..], g-cols [516+j0..]
        for (int i = tid; i < 4096; i += 256) {
            int kp = i >> 4, N = i & 15;
            int k = kp * 2;
            int colA = j0 + N;                       // <= 527 < 1032: safe
            BsA[N * BS_STRIDE_U32 + kp] =
                packbf(W2p[(long long)k * TWOOUT + colA],
                       W2p[(long long)(k + 1) * TWOOUT + colA]);
            int colG = OUT_ + j0 + N;
            BsG[N * BS_STRIDE_U32 + kp] = (colG < TWOOUT)
                   ? packbf(W2p[(long long)k * TWOOUT + colG],
                            W2p[(long long)(k + 1) * TWOOUT + colG])
                   : 0u;
        }
        __syncthreads();

        // prefetch next tile's W2 columns while this tile computes
        if (jt + 1 < 33) {
            int nj0 = (jt + 1) * 16;
            int kp = tid >> 4, N = tid & 15;        // 16 k-rows x 16 cols
            int k = kp * 32;                        // spread across K
            __builtin_prefetch(&W2p[(long long)k * TWOOUT + nj0 + N], 0, 1);
            int colG = OUT_ + nj0 + N;
            if (colG < TWOOUT)
                __builtin_prefetch(&W2p[(long long)k * TWOOUT + colG], 0, 1);
        }

        v8f accA = {}, accG = {};
        for (int k0 = 0; k0 < MLP; k0 += 32) {
            int pb = (k0 >> 1) + half * 4;
            Frag aF, bA, bG;
            load_frag(aF, &Hu[(m0 + l16) * (H_STRIDE_H / 2) + pb]);
            load_frag(bA, &BsA[l16 * BS_STRIDE_U32 + pb]);
            load_frag(bG, &BsG[l16 * BS_STRIDE_U32 + pb]);
            accA = __builtin_amdgcn_wmma_f32_16x16x32_bf16(
                false, aF.v, false, bA.v, (short)0, accA, false, false);
            accG = __builtin_amdgcn_wmma_f32_16x16x32_bf16(
                false, aF.v, false, bG.v, (short)0, accG, false, false);
        }
        int col = j0 + l16;
        if (col < OUT_) {
            float biasA = b2p[col];
            float biasG = b2p[OUT_ + col];
            int c    = col / 258;                    // C index (BW*REIM = 258)
            int rem  = col - c * 258;
            int bw   = rem >> 1;
            int reim = rem & 1;
            float wgt = fwp[bw];
            int f = HOP * n + bw;
            #pragma unroll
            for (int r = 0; r < 8; ++r) {
                int m = m0 + r + half * 8;
                int t = t0 + m;
                float a = accA[r] + biasA;
                float g = accG[r] + biasG;
                float mv = a * fast_sigmoid(g) * wgt;
                long long oidx = ((((long long)b * 2 + c) * N_FREQ + f) * 2048 + t) * 2 + reim;
                unsafeAtomicAdd(outp + oidx, mv);    // global_atomic_add_f32
            }
        }
    }
}

extern "C" void kernel_launch(void* const* d_in, const int* in_sizes, int n_in,
                              void* d_out, int out_size, void* d_ws, size_t ws_size,
                              hipStream_t stream) {
    const float* q   = (const float*)d_in[0];
    const float* lg  = (const float*)d_in[1];
    const float* lb  = (const float*)d_in[2];
    const float* W1  = (const float*)d_in[3];
    const float* b1  = (const float*)d_in[4];
    const float* W2  = (const float*)d_in[5];
    const float* b2  = (const float*)d_in[6];
    const float* fw  = (const float*)d_in[7];
    float* outp = (float*)d_out;

    (void)d_ws; (void)ws_size; (void)in_sizes; (void)n_in;

    // allow >64KB dynamic LDS
    hipFuncSetAttribute((const void*)band_mlp_mask_kernel,
                        hipFuncAttributeMaxDynamicSharedMemorySize, SMEM_BYTES);

    zero_f32<<<2048, 256, 0, stream>>>(outp, (long long)out_size);

    dim3 grid(N_BANDS * 2 * (2048 / TTILE));   // 15 bands * 2 batch * 16 t-tiles = 480
    band_mlp_mask_kernel<<<grid, 256, SMEM_BYTES, stream>>>(
        q, lg, lb, W1, b1, W2, b2, fw, outp);
}
